// W8SDLinear_65730179498162
// MI455X (gfx1250) — compile-verified, matrix operand order checked by reading, not run on hardware
//
#include <hip/hip_runtime.h>
#include <hip/hip_bf16.h>
#include <stdint.h>

// ---------------------------------------------------------------------------
// Types matching CDNA5 WMMA builtin signatures
// ---------------------------------------------------------------------------
typedef int          v8i   __attribute__((ext_vector_type(8)));
typedef float        v8f   __attribute__((ext_vector_type(8)));
typedef float        v2f   __attribute__((ext_vector_type(2)));
typedef unsigned int u32x4 __attribute__((ext_vector_type(4)));
typedef int          i32x8 __attribute__((ext_vector_type(8)));
typedef int          i32x4 __attribute__((ext_vector_type(4)));

#define ALPHA_PAD 32      // storage stride for outlier slices (alpha == 32)
#define TM 128            // M tile per workgroup
#define TN 128            // N tile per workgroup
#define TK 64             // K step (== WMMA iu8 K)
#define LDS_STRIDE 72     // 64B of K data + 8B TDM pad per row (bank skew)

#if __has_builtin(__builtin_amdgcn_tensor_load_to_lds) && \
    __has_builtin(__builtin_amdgcn_s_wait_tensorcnt)
#define USE_TDM 1
#else
#define USE_TDM 0
#endif

#if __has_builtin(__builtin_amdgcn_wmma_f32_16x16x4_f32)
#define USE_WMMA_F32K4 1
#else
#define USE_WMMA_F32K4 0
#endif

// ---------------------------------------------------------------------------
// Kernel 1: top-alpha outlier channels of |act_max| + channel mask
// ---------------------------------------------------------------------------
__global__ __launch_bounds__(256) void topk_mask_kernel(
    const float* __restrict__ act_max, const int* __restrict__ alpha_p,
    int* __restrict__ idx, unsigned char* __restrict__ mask, int K)
{
  __shared__ float sv[4096];
  __shared__ float rv[256];
  __shared__ int   ri[256];
  const int tid = threadIdx.x;
  const int na  = *alpha_p;
  for (int k = tid; k < K; k += 256) { sv[k] = fabsf(act_max[k]); mask[k] = 1; }
  __syncthreads();
  for (int it = 0; it < na; ++it) {
    float bv = -1.0f; int bi = K;
    for (int k = tid; k < K; k += 256) {
      float v = sv[k];
      if (v > bv) { bv = v; bi = k; }           // ascending scan => smallest idx on tie
    }
    rv[tid] = bv; ri[tid] = bi;
    __syncthreads();
    for (int s = 128; s > 0; s >>= 1) {
      if (tid < s) {
        float ov = rv[tid + s]; int oi = ri[tid + s];
        if (ov > rv[tid] || (ov == rv[tid] && oi < ri[tid])) { rv[tid] = ov; ri[tid] = oi; }
      }
      __syncthreads();
    }
    if (tid == 0) { idx[it] = ri[0]; mask[ri[0]] = 0; sv[ri[0]] = -2.0f; }
    __syncthreads();
  }
}

// ---------------------------------------------------------------------------
// Kernel 2: per-row masked int8 quantization + outlier gather.
// Used for both weight rows ([N,K]) and token rows ([M,K]).
// scale = fp16(maxabs/127); q = trunc(v/scale); outl[r][a] = src[r][idx[a]]
// ---------------------------------------------------------------------------
__global__ __launch_bounds__(256) void quant_rows_kernel(
    const float* __restrict__ src, const unsigned char* __restrict__ mask,
    const int* __restrict__ idx, const int* __restrict__ alpha_p,
    signed char* __restrict__ q, float* __restrict__ qmax,
    float* __restrict__ outl, int K)
{
  __shared__ float row[4096];
  __shared__ float red[256];
  const int r = blockIdx.x, tid = threadIdx.x;
  const float* sr = src + (size_t)r * K;
  float mx = 0.0f;
  for (int k = tid; k < K; k += 256) {
    float v = mask[k] ? sr[k] : 0.0f;
    row[k] = v;
    mx = fmaxf(mx, fabsf(v));
  }
  red[tid] = mx;
  __syncthreads();
  for (int s = 128; s > 0; s >>= 1) {
    if (tid < s) red[tid] = fmaxf(red[tid], red[tid + s]);
    __syncthreads();
  }
  const float scale = (float)(_Float16)(red[0] * (1.0f / 127.0f));  // fp16 rounding
  const float inv   = (scale != 0.0f) ? 1.0f / scale : 0.0f;
  for (int k = tid; k < K; k += 256)
    q[(size_t)r * K + k] = (signed char)(int)(row[k] * inv);        // trunc toward 0
  if (tid == 0) qmax[r] = scale;
  const int na = *alpha_p;
  if (tid < na) outl[(size_t)r * ALPHA_PAD + tid] = sr[idx[tid]];   // unmasked gather
}

// ---------------------------------------------------------------------------
// TDM: 2-D tile (rows x 64B of K) -> LDS, row-major with 8B pad every 64B row.
// D# built per CDNA5 ISA ch.8 (group0 128b, group1 256b; groups 2/3 zero).
// ---------------------------------------------------------------------------
#if USE_TDM
__device__ __forceinline__ void tdm_load_tile(
    const signed char* gptr, unsigned lds_off,
    unsigned tensor_d0, unsigned tensor_d1,
    unsigned tile_d0, unsigned tile_d1, unsigned long long stride0)
{
  const unsigned long long ga = (unsigned long long)(uintptr_t)gptr;
  u32x4 g0;
  g0[0] = 1u;                                        // count=1 (valid descriptor)
  g0[1] = lds_off;                                   // lds_addr (bytes)
  g0[2] = (unsigned)(ga & 0xffffffffu);              // global_addr[31:0]
  g0[3] = (unsigned)((ga >> 32) & 0x01ffffffu)       // global_addr[56:32]
        | (2u << 30);                                // type=2 ("image")
  i32x8 g1;
  // data_size=0 (1B) | pad_enable | pad_interval=3 (16 dwords) | pad_amount=1 (2 dwords)
  g1[0] = (int)((1u << 20) | (3u << 22) | (1u << 25));
  g1[1] = (int)((tensor_d0 & 0xffffu) << 16);                              // dim0[15:0]
  g1[2] = (int)(((tensor_d0 >> 16) & 0xffffu) | ((tensor_d1 & 0xffffu) << 16));
  g1[3] = (int)(((tensor_d1 >> 16) & 0xffffu) | ((tile_d0 & 0xffffu) << 16));
  g1[4] = (int)(tile_d1 & 0xffffu);                  // tile_dim1, tile_dim2=0
  g1[5] = (int)(unsigned)(stride0 & 0xffffffffull);  // dim0_stride[31:0]
  g1[6] = (int)(unsigned)((stride0 >> 32) & 0xffffull);
  g1[7] = 0;
  const i32x4 z4 = {0, 0, 0, 0};
  const i32x8 z8 = {0, 0, 0, 0, 0, 0, 0, 0};
  // clang-23 / therock-10.0 form: (g0, g1, g2, g3, g4, cpol)
  __builtin_amdgcn_tensor_load_to_lds(g0, g1, z4, z4, z8, 0);
}
#endif

// Load one 16x64 int8 WMMA operand fragment from an LDS tile.
// Per ISA 8-bit A layout: lane half g holds K chunks at bytes g*8 + {0,16,32,48}.
__device__ __forceinline__ v8i load_frag8(const signed char* rowbase, int half)
{
  const unsigned long long* p = (const unsigned long long*)rowbase;  // row is 8B-aligned (72B stride)
  unsigned long long d0 = p[half + 0];
  unsigned long long d1 = p[half + 2];
  unsigned long long d2 = p[half + 4];
  unsigned long long d3 = p[half + 6];
  v8i f;
  f[0] = (int)d0; f[1] = (int)(d0 >> 32);
  f[2] = (int)d1; f[3] = (int)(d1 >> 32);
  f[4] = (int)d2; f[5] = (int)(d2 >> 32);
  f[6] = (int)d3; f[7] = (int)(d3 >> 32);
  return f;
}

#if !USE_TDM
// Fallback cooperative copy: 128 rows x 64B, dst stride 72B.
__device__ __forceinline__ void copy_tile(signed char* dst, const signed char* src, int ldsrc)
{
  const int t = threadIdx.x, row = t >> 1, seg = t & 1;
  const int4* s = (const int4*)(src + (size_t)row * ldsrc + seg * 32);
  int4* d = (int4*)(dst + row * LDS_STRIDE + seg * 32);
  d[0] = s[0]; d[1] = s[1];
}
#endif

// ---------------------------------------------------------------------------
// Kernel 3: main GEMM. out[m][n] = i32dot8(xq[m],wq[n]) * fp16(xmax[m]*wmax[n])
//                                 + dot32(xout[m], wout[n]) + bias[n]
// 128x128 tile / block; 8 waves (2x4); each wave 64x32 via 4x2 iu8 WMMA frags.
// ---------------------------------------------------------------------------
__global__ __launch_bounds__(256) void w8sd_gemm_kernel(
    const signed char* __restrict__ xq, const signed char* __restrict__ wq,
    const float* __restrict__ xmax, const float* __restrict__ wmax,
    const float* __restrict__ xout, const float* __restrict__ wout,
    const float* __restrict__ bias, const int* __restrict__ alpha_p,
    float* __restrict__ out, int M, int N, int K)
{
  __shared__ __align__(16) signed char ldsA[2][TM * LDS_STRIDE];
  __shared__ __align__(16) signed char ldsB[2][TN * LDS_STRIDE];

  const int tile_m = blockIdx.y * TM;
  const int tile_n = blockIdx.x * TN;
  const int wid  = threadIdx.x >> 5;
  const int lid  = threadIdx.x & 31;
  const int wm   = wid >> 2;          // 0..1  (64 rows each)
  const int wn   = wid & 3;           // 0..3  (32 cols each)
  const int col  = lid & 15;
  const int half = lid >> 4;

  v8i acc[4][2];
  const v8i zero8 = {0, 0, 0, 0, 0, 0, 0, 0};
#pragma unroll
  for (int fm = 0; fm < 4; ++fm)
#pragma unroll
    for (int fn = 0; fn < 2; ++fn) acc[fm][fn] = zero8;

  const int KT = K / TK;

#if USE_TDM
  if (wid == 0) {
    tdm_load_tile(xq + (size_t)tile_m * K, (unsigned)(uintptr_t)(void*)&ldsA[0][0],
                  (unsigned)K, (unsigned)M, TK, TM, (unsigned long long)K);
    tdm_load_tile(wq + (size_t)tile_n * K, (unsigned)(uintptr_t)(void*)&ldsB[0][0],
                  (unsigned)K, (unsigned)N, TK, TN, (unsigned long long)K);
  }
  for (int kt = 0; kt < KT; ++kt) {
    const int buf = kt & 1;
    if (wid == 0) {
      if (kt + 1 < KT) {
        tdm_load_tile(xq + (size_t)tile_m * K + (size_t)(kt + 1) * TK,
                      (unsigned)(uintptr_t)(void*)&ldsA[buf ^ 1][0],
                      (unsigned)K, (unsigned)M, TK, TM, (unsigned long long)K);
        tdm_load_tile(wq + (size_t)tile_n * K + (size_t)(kt + 1) * TK,
                      (unsigned)(uintptr_t)(void*)&ldsB[buf ^ 1][0],
                      (unsigned)K, (unsigned)N, TK, TN, (unsigned long long)K);
        __builtin_amdgcn_s_wait_tensorcnt(2);   // current tile's pair retired
      } else {
        __builtin_amdgcn_s_wait_tensorcnt(0);
      }
    }
    __syncthreads();
    const signed char* At = &ldsA[buf][0];
    const signed char* Bt = &ldsB[buf][0];
#else
  for (int kt = 0; kt < KT; ++kt) {
    __syncthreads();
    copy_tile(&ldsA[0][0], xq + (size_t)tile_m * K + (size_t)kt * TK, K);
    copy_tile(&ldsB[0][0], wq + (size_t)tile_n * K + (size_t)kt * TK, K);
    __syncthreads();
    const signed char* At = &ldsA[0][0];
    const signed char* Bt = &ldsB[0][0];
#endif
    v8i afr[4], bfr[2];
#pragma unroll
    for (int fm = 0; fm < 4; ++fm)
      afr[fm] = load_frag8(At + (wm * 64 + fm * 16 + col) * LDS_STRIDE, half);
#pragma unroll
    for (int fn = 0; fn < 2; ++fn)
      bfr[fn] = load_frag8(Bt + (wn * 32 + fn * 16 + col) * LDS_STRIDE, half);
#pragma unroll
    for (int fm = 0; fm < 4; ++fm)
#pragma unroll
      for (int fn = 0; fn < 2; ++fn)
        acc[fm][fn] = __builtin_amdgcn_wmma_i32_16x16x64_iu8(
            true, afr[fm], true, bfr[fn], acc[fm][fn], false, false);
#if USE_TDM
    __syncthreads();   // all reads done before TDM overwrites this buffer
#endif
  }

  // -------------------------- epilogue --------------------------
  const int na = *alpha_p;
#pragma unroll
  for (int fm = 0; fm < 4; ++fm) {
#pragma unroll
    for (int fn = 0; fn < 2; ++fn) {
      const int n  = tile_n + wn * 32 + fn * 16 + col;
      const float bs  = bias[n];
      const float wmx = wmax[n];
      v8f facc;
#pragma unroll
      for (int j = 0; j < 8; ++j) {
        const int m = tile_m + wm * 64 + fm * 16 + half * 8 + j;
        const _Float16 h = (_Float16)xmax[m] * (_Float16)wmx;   // fp16 product, like ref
        facc[j] = (float)acc[fm][fn][j] * (float)h;
      }
#if USE_WMMA_F32K4
      // Outlier GEMM (K = alpha) folded in as chained f32 16x16x4 WMMA.
      const int arow = tile_m + wm * 64 + fm * 16 + col;        // A-frag M index = col
      for (int k = 0; k < na; k += 4) {
        const int kk = k + 2 * half;                            // lanes 16-31 hold K+2/K+3
        v2f a, b;
        a[0] = xout[(size_t)arow * ALPHA_PAD + kk];
        a[1] = xout[(size_t)arow * ALPHA_PAD + kk + 1];
        b[0] = wout[(size_t)n * ALPHA_PAD + kk];
        b[1] = wout[(size_t)n * ALPHA_PAD + kk + 1];
        facc = __builtin_amdgcn_wmma_f32_16x16x4_f32(
            false, a, false, b, (short)0, facc, false, false);
      }
#else
#pragma unroll
      for (int j = 0; j < 8; ++j) {
        const int m = tile_m + wm * 64 + fm * 16 + half * 8 + j;
        float s = 0.0f;
        for (int a = 0; a < na; ++a)
          s += xout[(size_t)m * ALPHA_PAD + a] * wout[(size_t)n * ALPHA_PAD + a];
        facc[j] += s;
      }
#endif
#pragma unroll
      for (int j = 0; j < 8; ++j) {
        const int m = tile_m + wm * 64 + fm * 16 + half * 8 + j;
        out[(size_t)m * N + n] = facc[j] + bs;
      }
    }
  }
}

// ---------------------------------------------------------------------------
// Host-side launcher
// ---------------------------------------------------------------------------
extern "C" void kernel_launch(void* const* d_in, const int* in_sizes, int n_in,
                              void* d_out, int out_size, void* d_ws, size_t ws_size,
                              hipStream_t stream) {
  const float* x       = (const float*)d_in[0];   // [M, K]
  const float* W       = (const float*)d_in[1];   // [N, K]
  const float* bias    = (const float*)d_in[2];   // [N]
  const float* act_max = (const float*)d_in[3];   // [K]
  const int*   alpha   = (const int*)d_in[4];     // scalar (32)

  const int K = in_sizes[3];
  const int N = in_sizes[2];
  const int M = in_sizes[0] / K;

  // ----- carve workspace (256B aligned slices) -----
  char* ws = (char*)d_ws;
  size_t off = 0;
  auto carve = [&](size_t bytes) {
    char* p = ws + off;
    off = (off + bytes + 255) & ~(size_t)255;
    return p;
  };
  int*           idx  = (int*)carve((size_t)ALPHA_PAD * sizeof(int));
  unsigned char* mask = (unsigned char*)carve((size_t)K);
  float*         wmx  = (float*)carve((size_t)N * sizeof(float));
  float*         xmx  = (float*)carve((size_t)M * sizeof(float));
  float*         wout = (float*)carve((size_t)N * ALPHA_PAD * sizeof(float));
  float*         xout = (float*)carve((size_t)M * ALPHA_PAD * sizeof(float));
  signed char*   wq   = (signed char*)carve((size_t)N * K);
  signed char*   xq   = (signed char*)carve((size_t)M * K);
  (void)ws_size; (void)n_in; (void)out_size;

  topk_mask_kernel<<<1, 256, 0, stream>>>(act_max, alpha, idx, mask, K);
  quant_rows_kernel<<<N, 256, 0, stream>>>(W, mask, idx, alpha, wq, wmx, wout, K);
  quant_rows_kernel<<<M, 256, 0, stream>>>(x, mask, idx, alpha, xq, xmx, xout, K);

  dim3 grid(N / TN, M / TM);
  w8sd_gemm_kernel<<<grid, 256, 0, stream>>>(xq, wq, xmx, wmx, xout, wout,
                                             bias, alpha, (float*)d_out, M, N, K);
}
